// MultiSSIMLoss_66855460929961
// MI455X (gfx1250) — compile-verified, matrix operand order checked by reading, not run on hardware
//
#include <hip/hip_runtime.h>
#include <hip/hip_bf16.h>

typedef __bf16 bf16;
typedef __attribute__((ext_vector_type(16))) __bf16 v16bf;
typedef __attribute__((ext_vector_type(8)))  float  v8f;

#define IMG_N 128
#define HW    256
#define PIX   65536
#define NCFG  25

__device__ __constant__ int   c_ks[5]  = {5, 7, 9, 11, 13};
__device__ __constant__ float c_sig[5] = {0.8f, 1.2f, 1.5f, 1.8f, 2.0f};

// Slot->K mapping for 16-bit WMMA operands (wave32), per CDNA5 ISA 7.12.2:
// lanes 0-15: VGPR0-3 hold K=0..7,  VGPR4-7 hold K=16..23
// lanes16-31: VGPR0-3 hold K=8..15, VGPR4-7 hold K=24..31 (two K per dword)
__device__ __forceinline__ int kOfSlot(int r, int pos, int hi) {
  int base = (r < 4) ? (2 * r) : (16 + 2 * (r - 4));
  return base + pos + 8 * hi;
}

union BP { unsigned u; bf16 h[2]; };

// ---------------- Stage 1: per-image sum / sumsq ----------------
__global__ __launch_bounds__(256)
void stats_kernel(const float* __restrict__ pred, const float* __restrict__ targ,
                  float2* __restrict__ stats) {
  const int which = blockIdx.x >> 7;      // 0 = pred, 1 = target
  const int img   = blockIdx.x & 127;
  const float* p = (which ? targ : pred) + (size_t)img * PIX;
  float s = 0.f, ss = 0.f;
  for (int i = threadIdx.x; i < PIX; i += 256) {
    float v = p[i];
    s += v; ss += v * v;
  }
  __shared__ float r0[256], r1[256];
  r0[threadIdx.x] = s; r1[threadIdx.x] = ss;
  __syncthreads();
  for (int st = 128; st > 0; st >>= 1) {
    if (threadIdx.x < st) {
      r0[threadIdx.x] += r0[threadIdx.x + st];
      r1[threadIdx.x] += r1[threadIdx.x + st];
    }
    __syncthreads();
  }
  if (threadIdx.x == 0) {
    float2 o; o.x = r0[0]; o.y = r1[0];
    stats[which * IMG_N + img] = o;
  }
}

// ---------------- Stage 2: standardize -> NP bf16 planes per image ----------------
// Plane order per image: x, y, [x*x, y*y, x*y]
template <int NP>
__global__ __launch_bounds__(256)
void standardize_kernel(const float* __restrict__ pred, const float* __restrict__ targ,
                        const float2* __restrict__ stats, bf16* __restrict__ planes) {
  const int img = blockIdx.y;
  const int i   = blockIdx.x * 256 + threadIdx.x;
  const float n = (float)PIX;
  float2 sp = stats[img];
  float2 st = stats[IMG_N + img];
  float mp = sp.x / n;
  float vp = (sp.y - n * mp * mp) / (n - 1.f);
  float ip = rsqrtf(fmaxf(vp, 1e-30f));
  float mt = st.x / n;
  float vt = (st.y - n * mt * mt) / (n - 1.f);
  float it = rsqrtf(fmaxf(vt, 1e-30f));
  const size_t off = (size_t)img * PIX + i;
  float xv = (pred[off] - mp) * ip;
  float yv = (targ[off] - mt) * it;
  const size_t base = (size_t)img * NP * PIX + i;
  planes[base]           = (bf16)xv;
  planes[base + PIX]     = (bf16)yv;
  if (NP == 5) {
    planes[base + 2 * (size_t)PIX] = (bf16)(xv * xv);
    planes[base + 3 * (size_t)PIX] = (bf16)(yv * yv);
    planes[base + 4 * (size_t)PIX] = (bf16)(xv * yv);
  }
}

__global__ void zero_kernel(float* __restrict__ p, int n) {
  int i = blockIdx.x * 256 + threadIdx.x;
  if (i < n) p[i] = 0.f;
}

// ---------------- Stage 3: tiled SSIM via bf16 WMMA ----------------
// grid: (256 tiles, 25 configs, 128 images); block: 1 wave (32 lanes)
template <int NP>
__global__ __launch_bounds__(32)
void ssim_tile_kernel(const bf16* __restrict__ planes, float* __restrict__ acc) {
  const int tile = blockIdx.x;
  const int cfg  = blockIdx.y;
  const int img  = blockIdx.z;
  const int tx = tile & 15, ty = tile >> 4;
  const int   ks    = c_ks[cfg / 5];
  const float sigma = c_sig[cfg % 5];
  const int pad = (ks - 1) >> 1;
  const int V   = HW - 2 * pad;          // valid output extent (244..252)
  const int lane = threadIdx.x;
  const int hi = lane >> 4, ml = lane & 15;

  __shared__ bf16     sQ[5][32][34];     // 5 quantity tiles; stride 34 -> conflict-free
  __shared__ unsigned sH32[16][17];      // staging, col-major, bf16 pairs packed in dwords
  __shared__ float    sG[16];

  // normalized 1D Gaussian weights in LDS
  {
    float g = 0.f;
    if (lane < ks) {
      float d = (float)lane - 0.5f * (float)(ks - 1);
      g = expf(-(d * d) / (2.f * sigma * sigma));
    }
    if (lane < 16) sG[lane] = g;
    __syncthreads();
    float gs = 0.f;
    for (int k = 0; k < ks; ++k) gs += sG[k];
    float gi = 1.f / gs;
    __syncthreads();
    if (lane < 16) sG[lane] = g * gi;
    __syncthreads();
  }

  // load 32x32 bf16 windows for all 5 quantities (zero-fill out of range)
  {
    const int r0 = 16 * ty, c0 = 16 * tx;
    const bf16* base = planes + (size_t)img * NP * PIX;
    const int r = r0 + lane;
    const bool interior = (r < HW) && (c0 + 32 <= HW);
    if (NP == 5) {
#pragma unroll
      for (int q = 0; q < 5; ++q) {
        const bf16* pq = base + (size_t)q * PIX;
        if (interior) {
          const uint4* p = (const uint4*)(pq + (size_t)r * HW + c0);  // 64B = 4x b128
          uint4 b[4];
#pragma unroll
          for (int t = 0; t < 4; ++t) b[t] = p[t];
          const bf16* e = (const bf16*)b;
#pragma unroll
          for (int j = 0; j < 32; ++j) sQ[q][lane][j] = e[j];
        } else {
          for (int j = 0; j < 32; ++j) {
            int c = c0 + j;
            bf16 v = (bf16)0.f;
            if (r < HW && c < HW) v = pq[(size_t)r * HW + c];
            sQ[q][lane][j] = v;
          }
        }
      }
    } else {
      // NP == 2 fallback: load x,y; derive products once per tile row
      if (interior) {
        const uint4* px = (const uint4*)(base + (size_t)r * HW + c0);
        const uint4* py = (const uint4*)(base + PIX + (size_t)r * HW + c0);
        uint4 bx[4], by[4];
#pragma unroll
        for (int t = 0; t < 4; ++t) { bx[t] = px[t]; by[t] = py[t]; }
        const bf16* ex = (const bf16*)bx;
        const bf16* ey = (const bf16*)by;
#pragma unroll
        for (int j = 0; j < 32; ++j) {
          float xv = (float)ex[j], yv = (float)ey[j];
          sQ[0][lane][j] = ex[j];
          sQ[1][lane][j] = ey[j];
          sQ[2][lane][j] = (bf16)(xv * xv);
          sQ[3][lane][j] = (bf16)(yv * yv);
          sQ[4][lane][j] = (bf16)(xv * yv);
        }
      } else {
        for (int j = 0; j < 32; ++j) {
          int c = c0 + j;
          float xv = 0.f, yv = 0.f;
          if (r < HW && c < HW) {
            xv = (float)base[(size_t)r * HW + c];
            yv = (float)base[PIX + (size_t)r * HW + c];
          }
          sQ[0][lane][j] = (bf16)xv;
          sQ[1][lane][j] = (bf16)yv;
          sQ[2][lane][j] = (bf16)(xv * xv);
          sQ[3][lane][j] = (bf16)(yv * yv);
          sQ[4][lane][j] = (bf16)(xv * yv);
        }
      }
    }
  }
  __syncthreads();

  // Banded-Gaussian fragment: slot value = g[k - (lane&15)].
  // Serves as horizontal B (32x16, B[k][n]=g[k-n]) and vertical A (16x32, A[m][k]=g[k-m]).
  v16bf Gfrag;
#pragma unroll
  for (int r = 0; r < 8; ++r)
#pragma unroll
    for (int pos = 0; pos < 2; ++pos) {
      int k = kOfSlot(r, pos, hi);
      int t = k - ml;
      float w = (t >= 0 && t < ks) ? sG[t] : 0.f;
      Gfrag[2 * r + pos] = (bf16)w;
    }

  v8f R[5];   // blurred mu_x, mu_y, E[x^2], E[y^2], E[xy]
#pragma unroll
  for (int q = 0; q < 5; ++q) {
    v8f h0, h1;
#pragma unroll
    for (int h = 0; h < 2; ++h) {       // input-row halves 0..15 / 16..31
      v16bf A;
      const unsigned* rowp = (const unsigned*)(&sQ[q][ml + 16 * h][0]);
#pragma unroll
      for (int r = 0; r < 8; ++r) {
        int kb = kOfSlot(r, 0, hi);     // even
        BP p; p.u = rowp[kb >> 1];
        A[2 * r]     = p.h[0];
        A[2 * r + 1] = p.h[1];
      }
      v8f z;
#pragma unroll
      for (int i = 0; i < 8; ++i) z[i] = 0.f;
      v8f d = __builtin_amdgcn_wmma_f32_16x16x32_bf16(
          false, A, false, Gfrag, (short)0, z, false, false);
      if (h == 0) h0 = d; else h1 = d;
    }
    __syncthreads();
    // stage row-blurred tile (32 rows x 16 cols), packed two rows per dword
#pragma unroll
    for (int r = 0; r < 8; r += 2) {
      BP p0; p0.h[0] = (bf16)h0[r]; p0.h[1] = (bf16)h0[r + 1];
      sH32[ml][(r + 8 * hi) >> 1] = p0.u;          // rows 0..15
      BP p1; p1.h[0] = (bf16)h1[r]; p1.h[1] = (bf16)h1[r + 1];
      sH32[ml][(16 + r + 8 * hi) >> 1] = p1.u;     // rows 16..31
    }
    __syncthreads();
    v16bf Bv;
#pragma unroll
    for (int r = 0; r < 8; ++r) {
      int kb = kOfSlot(r, 0, hi);       // even
      BP p; p.u = sH32[ml][kb >> 1];
      Bv[2 * r]     = p.h[0];
      Bv[2 * r + 1] = p.h[1];
    }
    v8f z2;
#pragma unroll
    for (int i = 0; i < 8; ++i) z2[i] = 0.f;
    R[q] = __builtin_amdgcn_wmma_f32_16x16x32_bf16(
        false, Gfrag, false, Bv, (short)0, z2, false, false);
    __syncthreads();
  }

  // elementwise SSIM over the 16x16 output tile (C/D layout: m = reg+8*hi, n = lane&15)
  const float C1v = 1e-4f, C2v = 9e-4f;
  float sum = 0.f;
#pragma unroll
  for (int r = 0; r < 8; ++r) {
    int m  = r + 8 * hi;
    int gm = 16 * ty + m, gn = 16 * tx + ml;
    if (gm < V && gn < V) {
      float mux = R[0][r], muy = R[1][r];
      float sxx = R[2][r] - mux * mux;
      float syy = R[3][r] - muy * muy;
      float sxy = R[4][r] - mux * muy;
      float num = (2.f * mux * muy + C1v) * (2.f * sxy + C2v);
      float den = (mux * mux + muy * muy + C1v) * (sxx + syy + C2v);
      sum += num / den;
    }
  }
#pragma unroll
  for (int off = 16; off > 0; off >>= 1)
    sum += __shfl_xor(sum, off, 32);
  if (lane == 0)
    atomicAdd(&acc[img * NCFG + cfg], sum / ((float)V * (float)V));
}

// ---------------- Stage 4: combine 128x25 -> scalar loss ----------------
__global__ __launch_bounds__(128)
void finalize_kernel(const float* __restrict__ acc, float* __restrict__ out) {
  const int i = threadIdx.x;     // image index
  float v = 0.f;
  for (int c = 0; c < NCFG; ++c) v += acc[i * NCFG + c];
  float loss = 0.5f - 0.5f * (v * (1.f / NCFG));  // 1 - (mean+1)/2
  __shared__ float red[128];
  red[i] = loss;
  __syncthreads();
  for (int s = 64; s > 0; s >>= 1) {
    if (i < s) red[i] += red[i + s];
    __syncthreads();
  }
  if (i == 0) out[0] = red[0] * (1.f / IMG_N);
}

extern "C" void kernel_launch(void* const* d_in, const int* in_sizes, int n_in,
                              void* d_out, int out_size, void* d_ws, size_t ws_size,
                              hipStream_t stream) {
  (void)in_sizes; (void)n_in; (void)out_size;
  const float* pred = (const float*)d_in[0];
  const float* targ = (const float*)d_in[1];
  char* ws = (char*)d_ws;
  float2* stats  = (float2*)ws;                    // 2 KB
  float*  acc    = (float*)(ws + 4096);            // 12.8 KB
  bf16*   planes = (bf16*)(ws + 32768);

  const size_t need5 = 32768 + (size_t)5 * IMG_N * PIX * 2;  // ~84 MB
  const bool precomp = (ws_size >= need5);

  hipLaunchKernelGGL(stats_kernel, dim3(2 * IMG_N), dim3(256), 0, stream,
                     pred, targ, stats);
  if (precomp) {
    hipLaunchKernelGGL((standardize_kernel<5>), dim3(PIX / 256, IMG_N), dim3(256), 0,
                       stream, pred, targ, stats, planes);
  } else {
    hipLaunchKernelGGL((standardize_kernel<2>), dim3(PIX / 256, IMG_N), dim3(256), 0,
                       stream, pred, targ, stats, planes);
  }
  hipLaunchKernelGGL(zero_kernel, dim3((IMG_N * NCFG + 255) / 256), dim3(256), 0, stream,
                     acc, IMG_N * NCFG);
  if (precomp) {
    hipLaunchKernelGGL((ssim_tile_kernel<5>), dim3(256, NCFG, IMG_N), dim3(32), 0,
                       stream, planes, acc);
  } else {
    hipLaunchKernelGGL((ssim_tile_kernel<2>), dim3(256, NCFG, IMG_N), dim3(32), 0,
                       stream, planes, acc);
  }
  hipLaunchKernelGGL(finalize_kernel, dim3(1), dim3(128), 0, stream,
                     acc, (float*)d_out);
}
